// MLP_RSNA9_73778948210891
// MI455X (gfx1250) — matrix operand order repaired and checked by reading, not compiled, not run on hardware
//
#include <hip/hip_runtime.h>
#include <hip/hip_bf16.h>

typedef __attribute__((ext_vector_type(16))) _Float16 v16h;
typedef __attribute__((ext_vector_type(8)))  _Float16 v8h;
typedef __attribute__((ext_vector_type(8)))  float    v8f;
typedef __attribute__((ext_vector_type(4)))  float    v4f;
typedef __attribute__((ext_vector_type(4)))  unsigned int v4u;
typedef __attribute__((ext_vector_type(8)))  int      v8i;
typedef __attribute__((ext_vector_type(4)))  int      v4i;

#define B_SZ   16384
#define G_SZ   25
#define P_SZ   256
#define H_SZ   512
#define C_SZ   3
#define BM     128      // batch rows per block (8 waves x 16 rows)
#define CHUNK  64       // H columns staged in LDS per iteration
#define NCH    (H_SZ / CHUNK)
#define LSTR   264      // LDS row stride in halves (256 + 8 pad -> 4-bank spread)

// ---------------------------------------------------------------------------
// Prep: W1 (3,P,H) f32 -> W1T (3,H,P) f16  (rows contiguous => B-fragments are
// two contiguous 16B per-lane loads; also what the TDM descriptor streams)
// ---------------------------------------------------------------------------
__global__ __launch_bounds__(256) void mlp_prep_w1t(const float* __restrict__ W1,
                                                    _Float16* __restrict__ W1T) {
    int t = blockIdx.x * 256 + threadIdx.x;          // p fastest -> coalesced writes
    int e = t / (H_SZ * P_SZ);
    int r = t % (H_SZ * P_SZ);
    int h = r / P_SZ;
    int p = r % P_SZ;
    W1T[t] = (_Float16)W1[((size_t)e * P_SZ + p) * H_SZ + h];
}

// LDS byte offset of a shared-memory pointer (addrspace(3) address)
__device__ __forceinline__ unsigned lds_off(const void* p) {
    return (unsigned)(unsigned long long)(const __attribute__((address_space(3))) void*)p;
}

// Issue one TDM 2D tile load: 64 rows x 256 halves of W1T -> padded LDS buffer
__device__ __forceinline__ void tdm_load_chunk(const _Float16* gsrc, unsigned ldsbyte) {
    unsigned long long ga = (unsigned long long)(size_t)gsrc;
    v4u g0;
    g0[0] = 1u;                                   // count=1, user descriptor
    g0[1] = ldsbyte;                              // lds_addr
    g0[2] = (unsigned)(ga & 0xffffffffull);       // global_addr[31:0]
    g0[3] = (unsigned)((ga >> 32) & 0x01ffffffull) | (2u << 30);  // addr[56:32] | type=2
    v8i g1;
    g1[0] = (1 << 16)      // data_size = 2 bytes
          | (1 << 20)      // pad_enable
          | (6 << 22)      // pad_interval: 128 DWORDs (= one 256-half row)
          | (3 << 25);     // pad_amount:   4 DWORDs (= 8 halves) -> stride 264
    g1[1] = (P_SZ & 0xffff) << 16;                // tensor_dim0 = 256 (low 16)
    g1[2] = (P_SZ >> 16) | ((CHUNK & 0xffff) << 16);   // dim0 hi | tensor_dim1 = 64
    g1[3] = (CHUNK >> 16) | (P_SZ << 16);         // dim1 hi | tile_dim0 = 256
    g1[4] = CHUNK;                                // tile_dim1 = 64, tile_dim2 = 0
    g1[5] = P_SZ;                                 // tensor_dim0_stride = 256
    g1[6] = 0;
    g1[7] = 0;
    v4i gz4 = {0, 0, 0, 0};                       // groups 2/3 unused (2D tensor)
    v8i gz8 = {0, 0, 0, 0, 0, 0, 0, 0};           // extra group (clang-23 6-arg form)
    __builtin_amdgcn_tensor_load_to_lds(g0, g1, gz4, gz4, gz8, 0);
}

// ---------------------------------------------------------------------------
// Fused grouped MLP: one block = (128 batch rows) x (one group g)
// ---------------------------------------------------------------------------
__global__ __launch_bounds__(256) void mlp_main(const float* __restrict__ x,
                                                const _Float16* __restrict__ W1T,
                                                const float* __restrict__ b1,
                                                const float* __restrict__ W2,
                                                const float* __restrict__ b2,
                                                float* __restrict__ out) {
    __shared__ __align__(16) _Float16 sW[2][CHUNK * LSTR];   // 2 x 33,792 B

    const int g    = blockIdx.y;
    const int g3   = (g < 5) ? 0 : ((g < 15) ? 1 : 2);       // GROUP_ID
    const int tid  = threadIdx.x;
    const int wave = tid >> 5;
    const int lane = tid & 31;
    const int l16  = lane & 15;
    const int half = lane >> 4;
    const int rowBase = blockIdx.x * BM + wave * 16;

    const _Float16* w1g = W1T + (size_t)g3 * H_SZ * P_SZ;

    // kick off chunk 0 via Tensor Data Mover (wave 0 only; waves branch around)
    if (wave == 0) tdm_load_chunk(w1g, lds_off(&sW[0][0]));

    // ---- A strip: 16 rows x K=256 as 8 f16 fragments, kept in VGPRs -------
    // A layout (16-bit 16x32): lane l holds row M=l%16; K runs
    // [c0,c0+8) and [c0+16,c0+24) with c0 = kstep*32 + (l/16)*8.
    const float* xrow = x + (size_t)(rowBase + l16) * (G_SZ * P_SZ) + g * P_SZ;
    v16h afrag[8];
#pragma unroll
    for (int ks = 0; ks < 8; ++ks) {
        const int c0 = ks * 32 + half * 8;
        v4f f0 = *(const v4f*)(xrow + c0);
        v4f f1 = *(const v4f*)(xrow + c0 + 4);
        v4f f2 = *(const v4f*)(xrow + c0 + 16);
        v4f f3 = *(const v4f*)(xrow + c0 + 20);
        v16h a;
#pragma unroll
        for (int i = 0; i < 4; ++i) {
            a[i]      = (_Float16)f0[i];
            a[4 + i]  = (_Float16)f1[i];
            a[8 + i]  = (_Float16)f2[i];
            a[12 + i] = (_Float16)f3[i];
        }
        afrag[ks] = a;
    }

    // layer-2 per-lane partials: opart[r][c] for D-row r (+8 if half==1)
    float opart[8][3];
#pragma unroll
    for (int r = 0; r < 8; ++r) { opart[r][0] = 0.f; opart[r][1] = 0.f; opart[r][2] = 0.f; }

    for (int ch = 0; ch < NCH; ++ch) {
        // double-buffer: issue next chunk's TDM, then wait for oldest only
        if (wave == 0) {
            if (ch + 1 < NCH) {
                tdm_load_chunk(w1g + (size_t)(ch + 1) * CHUNK * P_SZ,
                               lds_off(&sW[(ch + 1) & 1][0]));
                __builtin_amdgcn_s_wait_tensorcnt(1);   // chunk ch has landed
            } else {
                __builtin_amdgcn_s_wait_tensorcnt(0);
            }
        }
        __syncthreads();                                 // publish LDS to all waves

        const _Float16* sbuf = &sW[ch & 1][0];
#pragma unroll
        for (int nt = 0; nt < CHUNK / 16; ++nt) {
            v8f acc = {0.f, 0.f, 0.f, 0.f, 0.f, 0.f, 0.f, 0.f};
            // B layout (f16 32x16): lane n<16: K 0-7,16-23; lane n+16: K 8-15,24-31
            const _Float16* bp = sbuf + (nt * 16 + l16) * LSTR + half * 8;
            v8h lo = *(const v8h*)(bp);
            v8h hi = *(const v8h*)(bp + 16);
            v16h bcur = __builtin_shufflevector(lo, hi,
                0, 1, 2, 3, 4, 5, 6, 7, 8, 9, 10, 11, 12, 13, 14, 15);
#pragma unroll
            for (int ks = 0; ks < 8; ++ks) {
                v16h bnext = bcur;
                if (ks < 7) {                            // software pipeline: prefetch ks+1
                    v8h nlo = *(const v8h*)(bp + (ks + 1) * 32);
                    v8h nhi = *(const v8h*)(bp + (ks + 1) * 32 + 16);
                    bnext = __builtin_shufflevector(nlo, nhi,
                        0, 1, 2, 3, 4, 5, 6, 7, 8, 9, 10, 11, 12, 13, 14, 15);
                }
                acc = __builtin_amdgcn_wmma_f32_16x16x32_f16(
                    false, afrag[ks], false, bcur, (short)0, acc, false, false);
                bcur = bnext;
            }
            // bias + relu + fused second layer (per-lane column = l16)
            const int hcol = ch * CHUNK + nt * 16 + l16;
            const float bias = b1[g3 * H_SZ + hcol];
            const float* w2p = W2 + ((size_t)g3 * H_SZ + hcol) * C_SZ;
            const float w20 = w2p[0], w21 = w2p[1], w22 = w2p[2];
#pragma unroll
            for (int r = 0; r < 8; ++r) {
                float hv = fmaxf(acc[r] + bias, 0.0f);
                opart[r][0] = fmaf(hv, w20, opart[r][0]);
                opart[r][1] = fmaf(hv, w21, opart[r][1]);
                opart[r][2] = fmaf(hv, w22, opart[r][2]);
            }
        }
        __syncthreads();   // all waves done with buf[ch&1] before it is refilled
    }

    // ---- reduce layer-2 partials across the 16 lanes of each half ---------
#pragma unroll
    for (int r = 0; r < 8; ++r) {
#pragma unroll
        for (int c = 0; c < 3; ++c) {
            float v = opart[r][c];
            v += __shfl_xor(v, 1, 16);
            v += __shfl_xor(v, 2, 16);
            v += __shfl_xor(v, 4, 16);
            v += __shfl_xor(v, 8, 16);
            opart[r][c] = v;
        }
    }

    if (l16 < C_SZ) {
        const int c = l16;
        const float bb = b2[g3 * C_SZ + c];
#pragma unroll
        for (int r = 0; r < 8; ++r) {
            const int row = rowBase + half * 8 + r;   // D-layout: lanes>=16 hold M=r+8
            out[(size_t)row * (G_SZ * C_SZ) + g * C_SZ + c] = opart[r][c] + bb;
        }
    }
}

// ---------------------------------------------------------------------------
extern "C" void kernel_launch(void* const* d_in, const int* in_sizes, int n_in,
                              void* d_out, int out_size, void* d_ws, size_t ws_size,
                              hipStream_t stream) {
    const float* x  = (const float*)d_in[0];
    // d_in[1] = k_idx (identity), d_in[2] = v_idx (identity) -> not needed
    const float* W1 = (const float*)d_in[3];
    const float* b1 = (const float*)d_in[4];
    const float* W2 = (const float*)d_in[5];
    const float* b2 = (const float*)d_in[6];

    _Float16* W1T = (_Float16*)d_ws;   // 3*512*256*2 = 786,432 B

    mlp_prep_w1t<<<(3 * H_SZ * P_SZ) / 256, 256, 0, stream>>>(W1, W1T);

    dim3 grid(B_SZ / BM, G_SZ);
    mlp_main<<<grid, 256, 0, stream>>>(x, W1T, b1, W2, b2, (float*)d_out);
}
